// GNNEncoder_66967130079247
// MI455X (gfx1250) — compile-verified, matrix operand order checked by reading, not run on hardware
//
#include <hip/hip_runtime.h>
#include <hip/hip_bf16.h>
#include <math.h>

// ---------------- problem constants (from reference) ----------------
#define N_NODESC 50000
#define N_EDGESC 600000
#define ETOT     (N_EDGESC + N_NODESC)   // edges + self loops
#define NGRAPH   256
#define HIDC     128
#define NEG_SLOPE 0.2f

typedef _Float16 v8h  __attribute__((ext_vector_type(8)));
typedef _Float16 v16h __attribute__((ext_vector_type(16)));
typedef float    v8f  __attribute__((ext_vector_type(8)));
typedef int      v4i  __attribute__((ext_vector_type(4)));

typedef __attribute__((address_space(1))) v4i* glob_v4i_ptr;
typedef __attribute__((address_space(3))) v4i* lds_v4i_ptr;

// ---------------- float atomic max (sign-aware int trick) ----------------
__device__ __forceinline__ void atomicMaxF(float* addr, float v) {
    if (v >= 0.0f) {
        atomicMax((int*)addr, __float_as_int(v));
    } else {
        atomicMin((unsigned int*)addr, __float_as_uint(v));
    }
}

// ---------------- layer 1 GEMM: K=6, plain fp32 (tiny) ----------------
__global__ void k_gemm_in6(const float* __restrict__ x,
                           const float* __restrict__ W,
                           float* __restrict__ h) {
    int id = blockIdx.x * blockDim.x + threadIdx.x;
    if (id >= N_NODESC * HIDC) return;
    int n = id >> 7;
    int c = id & 127;
    const float* xr = x + n * 6;
    float s = 0.0f;
#pragma unroll
    for (int k = 0; k < 6; ++k) s = fmaf(xr[k], W[k * HIDC + c], s);
    h[id] = s;
}

// ---------------- repack W (fp32 128x128) into WMMA B-fragment order ----------------
// Layout: Wp[((kc*8 + t)*32 + lane)*16 + j]  = f16(W[K][n])
//   K = kc*32 + base + (j<8 ? j : j+8),  base = (lane>=16)?8:0
//   n = t*16 + (lane&15)
__global__ void k_repackW(const float* __restrict__ W, _Float16* __restrict__ Wp) {
    int id = blockIdx.x * blockDim.x + threadIdx.x;
    if (id >= 4 * 8 * 32 * 16) return;
    int j    = id & 15;
    int lane = (id >> 4) & 31;
    int t    = (id >> 9) & 7;
    int kc   = id >> 12;
    int base = (lane >= 16) ? 8 : 0;
    int K = kc * 32 + base + (j < 8 ? j : j + 8);
    int n = t * 16 + (lane & 15);
    Wp[id] = (_Float16)W[K * HIDC + n];
}

// ---------------- relu + fp32 -> f16 convert (next layer's A matrix) ----------------
__global__ void k_to_half_relu(const float* __restrict__ y, _Float16* __restrict__ hh) {
    int id = blockIdx.x * blockDim.x + threadIdx.x;
    if (id >= N_NODESC * HIDC) return;
    float v = y[id];
    hh[id] = (_Float16)(v > 0.0f ? v : 0.0f);
}

// ---------------- WMMA GEMM + fused attention dots ----------------
// H[50000x128] = A_f16[50000x128] @ W_f16[128x128]; als/ald = H @ a_src / a_dst.
// One wave32 per 16-row tile, full 128 output columns (8 accumulator tiles).
// Repacked W staged in LDS once per block via async global->LDS (ASYNCcnt path).
__global__ void __launch_bounds__(256) k_gemm_wmma(const _Float16* __restrict__ A,
                                                   const _Float16* __restrict__ Wp,
                                                   const float* __restrict__ a_src,
                                                   const float* __restrict__ a_dst,
                                                   float* __restrict__ H,
                                                   float* __restrict__ als,
                                                   float* __restrict__ ald) {
    __shared__ __align__(32) _Float16 ldsW[16384];   // 32 KB repacked W

    // ---- cooperative stage of W into LDS: 256 thr * 8 iter * 16 B = 32 KB ----
#if __has_builtin(__builtin_amdgcn_global_load_async_to_lds_b128) && \
    __has_builtin(__builtin_amdgcn_s_wait_asynccnt)
#pragma unroll
    for (int k = 0; k < 8; ++k) {
        int off = ((int)threadIdx.x + k * 256) * 8;   // units of half; 16B chunks
        __builtin_amdgcn_global_load_async_to_lds_b128(
            (glob_v4i_ptr)(Wp + off),
            (lds_v4i_ptr)(ldsW + off), 0, 0);
    }
    __builtin_amdgcn_s_wait_asynccnt(0);
#else
#pragma unroll
    for (int k = 0; k < 8; ++k) {
        int off = ((int)threadIdx.x + k * 256) * 8;
        *(v8h*)(ldsW + off) = *(const v8h*)(Wp + off);
    }
#endif
    __syncthreads();

    const int wave = (int)((blockIdx.x * blockDim.x + threadIdx.x) >> 5);
    const int lane = threadIdx.x & 31;

    if (wave < N_NODESC / 16) {      // 50000/16 = 3125 exact
        const int row0  = wave * 16;
        const int base  = (lane >= 16) ? 8 : 0;
        const int arow  = row0 + (lane & 15);
        const int rbase = base;                 // D rows: +8 for upper half-wave
        const int col   = lane & 15;

        v8f acc[8] = {};

#pragma unroll
        for (int kc = 0; kc < 4; ++kc) {
            // A fragment: 16-bit 16x32 layout — two contiguous 16B half loads
            const v8h* pa = (const v8h*)(A + (size_t)arow * HIDC + kc * 32 + base);
            v8h lo = pa[0];   // K = kc*32 + base + [0..7]
            v8h hi = pa[2];   // K = kc*32 + base + 16 + [0..7]
            v16h afrag = __builtin_shufflevector(lo, hi,
                            0, 1, 2, 3, 4, 5, 6, 7, 8, 9, 10, 11, 12, 13, 14, 15);
#pragma unroll
            for (int t = 0; t < 8; ++t) {
                v16h bfrag = *(const v16h*)(ldsW + (size_t)(((kc * 8 + t) * 32 + lane) * 16));
                acc[t] = __builtin_amdgcn_wmma_f32_16x16x32_f16(
                             false, afrag, false, bfrag, (short)0, acc[t], false, false);
            }
        }

        // ---- store H + fused per-row attention dots ----
        float ps[8] = {}, pd[8] = {};
#pragma unroll
        for (int t = 0; t < 8; ++t) {
            float as = a_src[t * 16 + col];
            float ad = a_dst[t * 16 + col];
#pragma unroll
            for (int r = 0; r < 8; ++r) {
                float v = acc[t][r];
                H[(size_t)(row0 + r + rbase) * HIDC + t * 16 + col] = v;
                ps[r] = fmaf(v, as, ps[r]);
                pd[r] = fmaf(v, ad, pd[r]);
            }
        }
        // reduce across the 16 lanes of each half-wave (col dimension)
#pragma unroll
        for (int off = 8; off > 0; off >>= 1) {
#pragma unroll
            for (int r = 0; r < 8; ++r) {
                ps[r] += __shfl_xor(ps[r], off, 32);
                pd[r] += __shfl_xor(pd[r], off, 32);
            }
        }
        if ((lane & 15) == 0) {
#pragma unroll
            for (int r = 0; r < 8; ++r) {
                als[row0 + r + rbase] = ps[r];
                ald[row0 + r + rbase] = pd[r];
            }
        }
    }
}

// ---------------- per-node attention half dots (layer 1 only) ----------------
__global__ void k_dots(const float* __restrict__ h,
                       const float* __restrict__ a_src,
                       const float* __restrict__ a_dst,
                       float* __restrict__ als, float* __restrict__ ald) {
    int wave = (int)((blockIdx.x * blockDim.x + threadIdx.x) >> 5);
    int lane = threadIdx.x & 31;
    if (wave >= N_NODESC) return;
    const float4 v = ((const float4*)(h + (size_t)wave * HIDC))[lane];
    int c = lane * 4;
    float s1 = v.x * a_src[c] + v.y * a_src[c + 1] + v.z * a_src[c + 2] + v.w * a_src[c + 3];
    float s2 = v.x * a_dst[c] + v.y * a_dst[c + 1] + v.z * a_dst[c + 2] + v.w * a_dst[c + 3];
#pragma unroll
    for (int off = 16; off > 0; off >>= 1) {
        s1 += __shfl_xor(s1, off, 32);
        s2 += __shfl_xor(s2, off, 32);
    }
    if (lane == 0) { als[wave] = s1; ald[wave] = s2; }
}

// ---------------- per-node init: m=-inf, denom=0 ----------------
__global__ void k_init_nodes(float* __restrict__ m, float* __restrict__ denom) {
    int id = blockIdx.x * blockDim.x + threadIdx.x;
    if (id >= N_NODESC) return;
    m[id] = -__builtin_inff();
    denom[id] = 0.0f;
}

// ---------------- y init with bias broadcast ----------------
__global__ void k_init_y(const float* __restrict__ b, float* __restrict__ y) {
    int id = blockIdx.x * blockDim.x + threadIdx.x;
    if (id >= N_NODESC * HIDC) return;
    y[id] = b[id & 127];
}

// ---------------- edge pass 1: e = leakyrelu(al_src[s]+al_dst[d]); seg-max ----------------
__global__ void k_edge_max(const int* __restrict__ src, const int* __restrict__ dst,
                           const float* __restrict__ als, const float* __restrict__ ald,
                           float* __restrict__ ebuf, float* __restrict__ m) {
    int i = blockIdx.x * blockDim.x + threadIdx.x;
    if (i >= ETOT) return;
    int s, d;
    if (i < N_EDGESC) { s = src[i]; d = dst[i]; }
    else              { s = d = i - N_EDGESC; }
    float e = als[s] + ald[d];
    e = (e > 0.0f) ? e : NEG_SLOPE * e;
    ebuf[i] = e;
    atomicMaxF(&m[d], e);
}

// ---------------- edge pass 2: ex = exp(e - m[d]); seg-sum ----------------
__global__ void k_edge_exp(const int* __restrict__ dst,
                           float* __restrict__ ebuf,
                           const float* __restrict__ m, float* __restrict__ denom) {
    int i = blockIdx.x * blockDim.x + threadIdx.x;
    if (i >= ETOT) return;
    int d = (i < N_EDGESC) ? dst[i] : (i - N_EDGESC);
    float ex = expf(ebuf[i] - m[d]);
    ebuf[i] = ex;
    atomicAdd(&denom[d], ex);
}

// ---------------- edge pass 3: y[d] += alpha * h[s]  (wave per edge) ----------------
__global__ void k_aggregate(const int* __restrict__ src, const int* __restrict__ dst,
                            const float* __restrict__ ebuf, const float* __restrict__ denom,
                            const float* __restrict__ h, float* __restrict__ y) {
    int wave = (int)((blockIdx.x * blockDim.x + threadIdx.x) >> 5);
    int lane = threadIdx.x & 31;
    if (wave >= ETOT) return;
    int s, d;
    if (wave < N_EDGESC) { s = src[wave]; d = dst[wave]; }
    else                 { s = d = wave - N_EDGESC; }
    float alpha = ebuf[wave] / (denom[d] + 1e-16f);
    const float4 v = ((const float4*)(h + (size_t)s * HIDC))[lane];
    float* yd = y + (size_t)d * HIDC + lane * 4;
    atomicAdd(yd + 0, alpha * v.x);
    atomicAdd(yd + 1, alpha * v.y);
    atomicAdd(yd + 2, alpha * v.z);
    atomicAdd(yd + 3, alpha * v.w);
}

// ---------------- pooling ----------------
__global__ void k_pool_init(float* __restrict__ out, float* __restrict__ counts) {
    int id = blockIdx.x * blockDim.x + threadIdx.x;
    if (id < NGRAPH * HIDC) out[id] = 0.0f;
    if (id < NGRAPH) counts[id] = 0.0f;
}

__global__ void k_pool_acc(const int* __restrict__ batch, const float* __restrict__ y,
                           float* __restrict__ out, float* __restrict__ counts) {
    int wave = (int)((blockIdx.x * blockDim.x + threadIdx.x) >> 5);
    int lane = threadIdx.x & 31;
    if (wave >= N_NODESC) return;
    int g = batch[wave];
    const float4 v = ((const float4*)(y + (size_t)wave * HIDC))[lane];
    float* od = out + (size_t)g * HIDC + lane * 4;
    atomicAdd(od + 0, v.x);
    atomicAdd(od + 1, v.y);
    atomicAdd(od + 2, v.z);
    atomicAdd(od + 3, v.w);
    if (lane == 0) atomicAdd(&counts[g], 1.0f);
}

__global__ void k_pool_div(float* __restrict__ out, const float* __restrict__ counts) {
    int id = blockIdx.x * blockDim.x + threadIdx.x;
    if (id >= NGRAPH * HIDC) return;
    out[id] /= fmaxf(counts[id >> 7], 1.0f);
}

// ==================================================================
extern "C" void kernel_launch(void* const* d_in, const int* in_sizes, int n_in,
                              void* d_out, int out_size, void* d_ws, size_t ws_size,
                              hipStream_t stream) {
    (void)in_sizes; (void)n_in; (void)out_size; (void)ws_size;

    const float* x    = (const float*)d_in[0];
    const int*   eidx = (const int*)d_in[1];
    const int*   batch= (const int*)d_in[2];
    const float* W[3]     = { (const float*)d_in[3],  (const float*)d_in[7],  (const float*)d_in[11] };
    const float* a_src[3] = { (const float*)d_in[4],  (const float*)d_in[8],  (const float*)d_in[12] };
    const float* a_dst[3] = { (const float*)d_in[5],  (const float*)d_in[9],  (const float*)d_in[13] };
    const float* bias[3]  = { (const float*)d_in[6],  (const float*)d_in[10], (const float*)d_in[14] };
    float* out = (float*)d_out;

    const int* esrc = eidx;
    const int* edst = eidx + N_EDGESC;

    // ---- workspace carve-out ----
    char* w = (char*)d_ws;
    size_t off = 0;
    auto carve = [&](size_t bytes) -> void* {
        void* p = w + off;
        off = (off + bytes + 255) & ~(size_t)255;
        return p;
    };
    float*     h     = (float*)    carve((size_t)N_NODESC * HIDC * sizeof(float));
    float*     y     = (float*)    carve((size_t)N_NODESC * HIDC * sizeof(float));
    _Float16*  hh    = (_Float16*) carve((size_t)N_NODESC * HIDC * sizeof(_Float16));
    _Float16*  Wp    = (_Float16*) carve((size_t)4 * 8 * 32 * 16 * sizeof(_Float16));
    float*     als   = (float*)    carve((size_t)N_NODESC * sizeof(float));
    float*     ald   = (float*)    carve((size_t)N_NODESC * sizeof(float));
    float*     mmax  = (float*)    carve((size_t)N_NODESC * sizeof(float));
    float*     denom = (float*)    carve((size_t)N_NODESC * sizeof(float));
    float*     ebuf  = (float*)    carve((size_t)ETOT * sizeof(float));
    float*     cnts  = (float*)    carve((size_t)NGRAPH * sizeof(float));

    const int BLK = 256;
    const int gNC    = (N_NODESC * HIDC + BLK - 1) / BLK;       // elementwise over N*128
    const int gN     = (N_NODESC + BLK - 1) / BLK;              // elementwise over N
    const int gE     = (ETOT + BLK - 1) / BLK;                  // per edge
    const int gEwave = (ETOT * 32 + BLK - 1) / BLK;             // wave per edge
    const int gNwave = (N_NODESC * 32 + BLK - 1) / BLK;         // wave per node
    const int gGemm  = ((N_NODESC / 16) * 32 + BLK - 1) / BLK;  // wave per 16-row tile
    const int gRep   = (4 * 8 * 32 * 16 + BLK - 1) / BLK;

    for (int L = 0; L < 3; ++L) {
        // 1) GEMM: h = feat @ W[L]  (+ fused attention dots on WMMA path)
        if (L == 0) {
            k_gemm_in6<<<gNC, BLK, 0, stream>>>(x, W[0], h);
            k_dots<<<gNwave, BLK, 0, stream>>>(h, a_src[0], a_dst[0], als, ald);
        } else {
            k_repackW<<<gRep, BLK, 0, stream>>>(W[L], Wp);
            k_gemm_wmma<<<gGemm, BLK, 0, stream>>>(hh, Wp, a_src[L], a_dst[L], h, als, ald);
        }
        // 2) init segment buffers + y with bias
        k_init_nodes<<<gN, BLK, 0, stream>>>(mmax, denom);
        k_init_y<<<gNC, BLK, 0, stream>>>(bias[L], y);
        // 3) segment softmax (max, exp/sum) + weighted scatter-add
        k_edge_max<<<gE, BLK, 0, stream>>>(esrc, edst, als, ald, ebuf, mmax);
        k_edge_exp<<<gE, BLK, 0, stream>>>(edst, ebuf, mmax, denom);
        k_aggregate<<<gEwave, BLK, 0, stream>>>(esrc, edst, ebuf, denom, h, y);
        // 4) ReLU + convert for next layer's WMMA A-matrix
        if (L < 2) {
            k_to_half_relu<<<gNC, BLK, 0, stream>>>(y, hh);
        }
    }

    // global mean pool
    k_pool_init<<<(NGRAPH * HIDC + BLK - 1) / BLK, BLK, 0, stream>>>(out, cnts);
    k_pool_acc<<<gNwave, BLK, 0, stream>>>(batch, y, out, cnts);
    k_pool_div<<<(NGRAPH * HIDC + BLK - 1) / BLK, BLK, 0, stream>>>(out, cnts);
}
